// SmallRIN_75892072121039
// MI455X (gfx1250) — matrix operand order repaired and checked by reading, not compiled, not run on hardware
//
#include <hip/hip_runtime.h>
#include <hip/hip_bf16.h>

#define PHI_F 1.6180339887498949f
#define Bb 8
#define Ss 128
#define Vv 32000
#define Dd 256
#define Nn 256
#define Ll 2
#define Mrows (Ss * Bb)   /* 1024 */
#define K2D   (2 * Dd)    /* 512 */

typedef float  v2f   __attribute__((ext_vector_type(2)));
typedef float  v8f   __attribute__((ext_vector_type(8)));
typedef __bf16 bf16_t;
typedef bf16_t v8bf  __attribute__((ext_vector_type(8)));
typedef bf16_t v16bf __attribute__((ext_vector_type(16)));

__device__ __forceinline__ float silu_f(float x) {
    return x / (1.0f + __expf(-x));
}

__device__ __forceinline__ v16bf load16bf(const bf16_t* p0, const bf16_t* p1) {
    v8bf a = *(const v8bf*)p0;
    v8bf b = *(const v8bf*)p1;
    return __builtin_shufflevector(a, b, 0, 1, 2, 3, 4, 5, 6, 7,
                                         8, 9, 10, 11, 12, 13, 14, 15);
}

// LDS byte offset of a __shared__ object: flat shared-aperture addresses carry
// the LDS offset in their low 32 bits (aperture base low bits are zero).
__device__ __forceinline__ uint32_t lds_addr32(const void* p) {
    return (uint32_t)(uintptr_t)p;
}

// 16-byte async copy global -> LDS (per-lane addresses), tracked by ASYNCcnt.
__device__ __forceinline__ void async_cp16(uint32_t lds_off, const void* g) {
    asm volatile("global_load_async_to_lds_b128 %0, %1, off"
                 :: "v"(lds_off), "v"((uint64_t)(uintptr_t)g)
                 : "memory");
}

__device__ __forceinline__ void wait_async0() {
    asm volatile("s_wait_asynccnt 0x0" ::: "memory");
}

// ---------------------------------------------------------------------------
// 1) Recurrence: 2048 independent chains (b,d), 128 serial steps.
//    Writes X[r, 0:256]=x_r, X[r, 256:512]=x_i with r = t*B + b.
// ---------------------------------------------------------------------------
__global__ void recurrence_kernel(const int* __restrict__ ids,
                                  const float* __restrict__ emb,
                                  float* __restrict__ X) {
    int tid = blockIdx.x * blockDim.x + threadIdx.x;   // 0..2047
    int b = tid >> 8;
    int d = tid & 255;
    float h_r = 0.0f, h_i = 0.0f;
    for (int t = 0; t < Ss; ++t) {
        int token = ids[b * Ss + t];
        const float* e = emb + (size_t)token * K2D;
        float w  = e[d];
        float be = e[Dd + d];
        float wl = 1.0f + fabsf(w);
        float tphi = (float)t * PHI_F;
        float th_r = h_r / wl + be + tphi;
        float th_i = h_i / wl + be + tphi;
        // cr*ci - sr*si == cos(th_r+th_i); cr*si + sr*ci == sin(th_r+th_i)
        float th = th_r + th_i;
        float s, c;
        sincosf(th, &s, &c);   // precise path: seeds everything downstream
        h_r = c;
        h_i = s;
        int r = t * Bb + b;
        X[(size_t)r * K2D + d]      = h_r;
        X[(size_t)r * K2D + Dd + d] = h_i;
    }
}

// ---------------------------------------------------------------------------
// 2) fp32 -> bf16 hi/lo split (for bf16x3 emulated-fp32 WMMA GEMM)
// ---------------------------------------------------------------------------
__global__ void split_bf16_kernel(const float* __restrict__ src,
                                  bf16_t* __restrict__ hi,
                                  bf16_t* __restrict__ lo, int n) {
    int i = blockIdx.x * blockDim.x + threadIdx.x;
    if (i < n) {
        float x = src[i];
        bf16_t h = (bf16_t)x;
        hi[i] = h;
        lo[i] = (bf16_t)(x - (float)h);
    }
}

// ---------------------------------------------------------------------------
// 3) Generic fp32 WMMA GEMM (NT): C[M,N] = A[M,K] * W[N,K]^T
//    mode 0: out[row*ldo + col]         = acc + bias[col]
//    mode 1: out[row*ldo + ncol0 + col] += silu(acc)   (residual update)
//    One wave per block; one 16x16 tile; v_wmma_f32_16x16x4_f32 (exact fp32:
//    these GEMMs feed sin() arguments, precision-sensitive, FLOP-trivial).
// ---------------------------------------------------------------------------
__global__ void gemm_f32_nt_kernel(const float* __restrict__ A,
                                   const float* __restrict__ W,
                                   const float* __restrict__ bias,
                                   float* __restrict__ out,
                                   int K, int ldo, int ncol0, int mode) {
    int lane = threadIdx.x;
    int m0 = blockIdx.x * 16;
    int n0 = blockIdx.y * 16;
    int hf  = lane >> 4;      // 0: K pair {0,1}; 1: K pair {2,3}
    int idx = lane & 15;      // A row / W row / output col within tile

    const float* arow = A + (size_t)(m0 + idx) * K;
    const float* wrow = W + (size_t)(n0 + idx) * K;

    float cc[8];
#if __has_builtin(__builtin_amdgcn_wmma_f32_16x16x4_f32)
    v8f c = {};
#pragma unroll 4
    for (int k = 0; k < K; k += 4) {
        int ko = k + hf * 2;
        v2f a = *(const v2f*)(arow + ko);
        v2f b = *(const v2f*)(wrow + ko);
        c = __builtin_amdgcn_wmma_f32_16x16x4_f32(
                false, a, false, b, (short)0, c, false, false);
    }
#pragma unroll
    for (int v = 0; v < 8; ++v) cc[v] = c[v];
#else
#pragma unroll
    for (int v = 0; v < 8; ++v) cc[v] = 0.0f;
    for (int k = 0; k < K; ++k) {
        float wv = W[(size_t)(n0 + idx) * K + k];
#pragma unroll
        for (int v = 0; v < 8; ++v)
            cc[v] += A[(size_t)(m0 + hf * 8 + v) * K + k] * wv;
    }
#endif

    int col = n0 + idx;
#pragma unroll
    for (int v = 0; v < 8; ++v) {
        int row = m0 + hf * 8 + v;
        float acc = cc[v];
        float* p = out + (size_t)row * ldo + ncol0 + col;
        if (mode == 0) {
            *p = acc + bias[col];
        } else {
            *p = *p + silu_f(acc);
        }
    }
}

// ---------------------------------------------------------------------------
// 4) Resonator core: cos/sin_sum[r,n] = sum_d trig(xc[r,d]*iwl[n,d]+Bn[n,d]+t*phi)
//    Block = 16 rows x 16 resonators. XC/Bn tiles staged via async-to-LDS;
//    Wn goes through registers (needs the 1/(1+|w|) transform).
//    Row stride 260 floats: 16B-aligned segments + conflict-free banks.
// ---------------------------------------------------------------------------
#define TS_STRIDE 260
__global__ void theta_sums_kernel(const float* __restrict__ XC,
                                  const float* __restrict__ Wn,
                                  const float* __restrict__ Bn,
                                  float* __restrict__ cos_sum,
                                  float* __restrict__ sin_sum) {
    __shared__ __align__(16) float xc_s[16][TS_STRIDE];
    __shared__ __align__(16) float iwl_s[16][TS_STRIDE];
    __shared__ __align__(16) float bn_s[16][TS_STRIDE];
    int tid = threadIdx.x;
    int r0 = blockIdx.x * 16;
    int n0 = blockIdx.y * 16;

    // async stage: 16 rows x 256 floats = 16KB per array = 4 issues x 256 lanes
#pragma unroll
    for (int i = 0; i < 4; ++i) {
        int f = i * 256 + tid;
        int row = f >> 6;          // 64 x 16B segments per row
        int seg = f & 63;
        async_cp16(lds_addr32(&xc_s[row][seg * 4]),
                   XC + (size_t)(r0 + row) * Dd + seg * 4);
        async_cp16(lds_addr32(&bn_s[row][seg * 4]),
                   Bn + (size_t)(n0 + row) * Dd + seg * 4);
    }
    // register path for the transformed reciprocal table
    for (int i = tid; i < 16 * 256; i += 256) {
        int row = i >> 8;
        int col = i & 255;
        float w = Wn[(size_t)(n0 + row) * Dd + col];
        iwl_s[row][col] = 1.0f / (1.0f + fabsf(w));
    }
    wait_async0();
    __syncthreads();

    int rl = tid & 15;
    int nl = tid >> 4;
    int r = r0 + rl;
    float tphi = (float)(r >> 3) * PHI_F;   // r = t*8 + b -> t = r>>3
    float cs = 0.0f, ss = 0.0f;
#pragma unroll 4
    for (int d = 0; d < Dd; ++d) {
        float th = fmaf(xc_s[rl][d], iwl_s[nl][d], bn_s[nl][d]) + tphi;
        cs += __cosf(th);
        ss += __sinf(th);
    }
    cos_sum[(size_t)r * Nn + n0 + nl] = cs;
    sin_sum[(size_t)r * Nn + n0 + nl] = ss;
}

// ---------------------------------------------------------------------------
// 5) Logits: bf16x3 (emulated fp32) WMMA GEMM with double-buffered
//    async-to-LDS staging. Block = 8 waves = 64x128 output tile; each wave a
//    16x64 sub-tile (12 v_wmma_f32_16x16x32_bf16 per 32-K chunk).
//    LDS row stride 40 elems -> banks 20*r mod 64: conflict-free b128 reads.
// ---------------------------------------------------------------------------
#define KCH 32
#define LSTR 40
__global__ __launch_bounds__(256)
void logits_kernel(const bf16_t* __restrict__ Xhi,
                   const bf16_t* __restrict__ Xlo,
                   const bf16_t* __restrict__ Whi,
                   const bf16_t* __restrict__ Wlo,
                   const float* __restrict__ outb,
                   float* __restrict__ out) {
    __shared__ __align__(16) bf16_t Ah[2][64 * LSTR];
    __shared__ __align__(16) bf16_t Al[2][64 * LSTR];
    __shared__ __align__(16) bf16_t Bh[2][128 * LSTR];
    __shared__ __align__(16) bf16_t Bl[2][128 * LSTR];

    const int tid = threadIdx.x;
    const int m0 = blockIdx.x * 64;
    const int n0 = blockIdx.y * 128;

    // copy mapping: each lane moves one 16B segment (8 bf16) per issue;
    // a 32-K row = 4 segments -> 4 lanes per row.
    const int crow = tid >> 2;    // 0..63
    const int cseg = tid & 3;     // 0..3
    const uint32_t ldsElem = (uint32_t)(crow * LSTR + cseg * 8);
    const bf16_t* gAh  = Xhi + (size_t)(m0 + crow) * K2D + cseg * 8;
    const bf16_t* gAl  = Xlo + (size_t)(m0 + crow) * K2D + cseg * 8;
    const bf16_t* gBh0 = Whi + (size_t)(n0 + crow) * K2D + cseg * 8;
    const bf16_t* gBl0 = Wlo + (size_t)(n0 + crow) * K2D + cseg * 8;
    const bf16_t* gBh1 = Whi + (size_t)(n0 + 64 + crow) * K2D + cseg * 8;
    const bf16_t* gBl1 = Wlo + (size_t)(n0 + 64 + crow) * K2D + cseg * 8;

    auto issue = [&](int buf, int k0) {
        async_cp16(lds_addr32(&Ah[buf][ldsElem]), gAh + k0);
        async_cp16(lds_addr32(&Al[buf][ldsElem]), gAl + k0);
        async_cp16(lds_addr32(&Bh[buf][ldsElem]), gBh0 + k0);
        async_cp16(lds_addr32(&Bh[buf][64 * LSTR + ldsElem]), gBh1 + k0);
        async_cp16(lds_addr32(&Bl[buf][ldsElem]), gBl0 + k0);
        async_cp16(lds_addr32(&Bl[buf][64 * LSTR + ldsElem]), gBl1 + k0);
    };

    const int w    = tid >> 5;
    const int lane = tid & 31;
    const int mi = w >> 1;        // 0..3: 16-row sub-band
    const int ni = w & 1;         // 0..1: 64-col sub-range
    const int hf  = lane >> 4;
    const int idx = lane & 15;
    const int arow = mi * 16 + idx;

    v8f z = {};
    v8f acc0 = z, acc1 = z, acc2 = z, acc3 = z;
    v8f* acc[4] = {&acc0, &acc1, &acc2, &acc3};

    issue(0, 0);
    for (int kc = 0; kc < K2D / KCH; ++kc) {
        const int buf = kc & 1;
        wait_async0();
        __syncthreads();
        if (kc + 1 < K2D / KCH) issue(buf ^ 1, (kc + 1) * KCH);

        // A fragment: K = {8h..8h+7, 16+8h..16+8h+7} within the chunk
        v16bf ahi = load16bf(&Ah[buf][arow * LSTR + hf * 8],
                             &Ah[buf][arow * LSTR + 16 + hf * 8]);
        v16bf alo = load16bf(&Al[buf][arow * LSTR + hf * 8],
                             &Al[buf][arow * LSTR + 16 + hf * 8]);
#pragma unroll
        for (int j = 0; j < 4; ++j) {
            int brow = ni * 64 + j * 16 + idx;   // B column, K = {8h..8h+7, 16+8h..16+8h+7}
            v16bf bhi = load16bf(&Bh[buf][brow * LSTR + hf * 8],
                                 &Bh[buf][brow * LSTR + 16 + hf * 8]);
            v16bf blo = load16bf(&Bl[buf][brow * LSTR + hf * 8],
                                 &Bl[buf][brow * LSTR + 16 + hf * 8]);
            v8f c = *acc[j];
            c = __builtin_amdgcn_wmma_f32_16x16x32_bf16(
                    false, ahi, false, bhi, (short)0, c, false, false);
            c = __builtin_amdgcn_wmma_f32_16x16x32_bf16(
                    false, ahi, false, blo, (short)0, c, false, false);
            c = __builtin_amdgcn_wmma_f32_16x16x32_bf16(
                    false, alo, false, bhi, (short)0, c, false, false);
            *acc[j] = c;
        }
        __syncthreads();
    }

#pragma unroll
    for (int j = 0; j < 4; ++j) {
        int col = n0 + ni * 64 + j * 16 + idx;
        float bv = outb[col];
        v8f c = *acc[j];
#pragma unroll
        for (int v = 0; v < 8; ++v) {
            int r = m0 + mi * 16 + hf * 8 + v;   // r = t*B + b
            int t = r >> 3;
            int b = r & 7;
            out[((size_t)(b * Ss + t)) * Vv + col] = c[v] + bv;
        }
    }
}

// ---------------------------------------------------------------------------
// Launcher
// ---------------------------------------------------------------------------
extern "C" void kernel_launch(void* const* d_in, const int* in_sizes, int n_in,
                              void* d_out, int out_size, void* d_ws, size_t ws_size,
                              hipStream_t stream) {
    (void)in_sizes; (void)n_in; (void)out_size; (void)ws_size;
    const int*   ids  = (const int*)d_in[0];
    const float* emb  = (const float*)d_in[1];
    const float* ip_w = (const float*)d_in[2];
    const float* ip_b = (const float*)d_in[3];
    const float* Wn   = (const float*)d_in[4];
    const float* Bn   = (const float*)d_in[5];
    const float* opr  = (const float*)d_in[6];
    const float* opi  = (const float*)d_in[7];
    const float* outw = (const float*)d_in[8];
    const float* outb = (const float*)d_in[9];
    float* out = (float*)d_out;

    char* ws = (char*)d_ws;
    float* X  = (float*)ws;  ws += (size_t)Mrows * K2D * 4;
    float* XC = (float*)ws;  ws += (size_t)Mrows * Dd * 4;
    float* CS = (float*)ws;  ws += (size_t)Mrows * Nn * 4;
    float* SSm = (float*)ws; ws += (size_t)Mrows * Nn * 4;
    bf16_t* Xhi = (bf16_t*)ws; ws += (size_t)Mrows * K2D * 2;
    bf16_t* Xlo = (bf16_t*)ws; ws += (size_t)Mrows * K2D * 2;
    bf16_t* Whi = (bf16_t*)ws; ws += (size_t)Vv * K2D * 2;
    bf16_t* Wlo = (bf16_t*)ws; ws += (size_t)Vv * K2D * 2;

    // 1) recurrence (serial in t, parallel over B*D)
    recurrence_kernel<<<8, 256, 0, stream>>>(ids, emb, X);

    // 2) split out_w -> bf16 hi/lo
    int nW = Vv * K2D;
    split_bf16_kernel<<<(nW + 255) / 256, 256, 0, stream>>>(outw, Whi, Wlo, nW);

    // 3) resonant layers, batched over all 1024 (t,b) rows
    for (int l = 0; l < Ll; ++l) {
        gemm_f32_nt_kernel<<<dim3(Mrows / 16, Dd / 16), 32, 0, stream>>>(
            X, ip_w + (size_t)l * Dd * K2D, ip_b + (size_t)l * Dd, XC,
            /*K=*/K2D, /*ldo=*/Dd, /*ncol0=*/0, /*mode=*/0);
        theta_sums_kernel<<<dim3(Mrows / 16, Nn / 16), 256, 0, stream>>>(
            XC, Wn + (size_t)l * Nn * Dd, Bn + (size_t)l * Nn * Dd, CS, SSm);
        gemm_f32_nt_kernel<<<dim3(Mrows / 16, Dd / 16), 32, 0, stream>>>(
            CS, opr + (size_t)l * Dd * Nn, nullptr, X,
            /*K=*/Nn, /*ldo=*/K2D, /*ncol0=*/0, /*mode=*/1);
        gemm_f32_nt_kernel<<<dim3(Mrows / 16, Dd / 16), 32, 0, stream>>>(
            SSm, opi + (size_t)l * Dd * Nn, nullptr, X,
            /*K=*/Nn, /*ldo=*/K2D, /*ncol0=*/Dd, /*mode=*/1);
    }

    // 4) split final X -> bf16 hi/lo
    int nX = Mrows * K2D;
    split_bf16_kernel<<<(nX + 255) / 256, 256, 0, stream>>>(X, Xhi, Xlo, nX);

    // 5) big logits GEMM (bf16x3 WMMA + async LDS double buffering)
    logits_kernel<<<dim3(Mrows / 64, Vv / 128), 256, 0, stream>>>(
        Xhi, Xlo, Whi, Wlo, outb, out);
}
